// RelationalTransformer_74646531604810
// MI455X (gfx1250) — matrix operand order repaired and checked by reading, not compiled, not run on hardware
//
#include <hip/hip_runtime.h>
#include <hip/hip_bf16.h>
#include <stdint.h>

// Model dims
#define B_   2
#define S_   1024
#define H_   12
#define D_   768
#define HD_  64
#define DFF_ 3072
#define DT_  384
#define L_   4
#define M_   (B_*S_)

typedef unsigned short u16t;
typedef unsigned char  u8t;
typedef unsigned int   u32t;
typedef __attribute__((ext_vector_type(4)))  int   i32x4;
typedef __attribute__((ext_vector_type(8)))  u16t  u16x8;
typedef __attribute__((ext_vector_type(16))) u16t  u16x16;
typedef __attribute__((ext_vector_type(16))) __bf16 bf16x16;
typedef __attribute__((ext_vector_type(8)))  float f32x8;

// CDNA5 async global->LDS copy path (ASYNCcnt), guarded so the sync
// fallback keeps the build green if the builtins are absent.
// Builtin signature (from hipcc diagnostic): (int4 AS1* src, int4 AS3* dst,
// imm offset, imm cpol).
#if defined(__gfx1250__) && \
    __has_builtin(__builtin_amdgcn_global_load_async_to_lds_b128) && \
    __has_builtin(__builtin_amdgcn_s_wait_asynccnt)
#define HAS_ASYNC_LDS 1
#define ASYNC_LOAD_B128(gsrc, ldst)                                        \
  __builtin_amdgcn_global_load_async_to_lds_b128(                          \
      (__attribute__((address_space(1))) i32x4*)(gsrc),                    \
      (__attribute__((address_space(3))) i32x4*)(ldst), 0, 0)
#define ASYNC_WAIT() __builtin_amdgcn_s_wait_asynccnt(0)
#else
#define HAS_ASYNC_LDS 0
#endif

__device__ __forceinline__ u16t f2bf(float f) {
  u32t u = __float_as_uint(f);
  u32t r = u + 0x7FFFu + ((u >> 16) & 1u);   // round-to-nearest-even
  return (u16t)(r >> 16);
}
__device__ __forceinline__ float bf2f(u16t h) {
  return __uint_as_float(((u32t)h) << 16);
}
__device__ __forceinline__ bf16x16 cat16(u16x8 lo, u16x8 hi) {
  u16x16 t;
#pragma unroll
  for (int i = 0; i < 8; ++i) { t[i] = lo[i]; t[8 + i] = hi[i]; }
  return __builtin_bit_cast(bf16x16, t);
}

// ---------------------------------------------------------------------------
// Tiled WMMA GEMM: C[M,N] = A[M,K] @ W[K,N] (+bias) (+residual)
// A: bf16 (or fp32 converted on load), W fp32 converted to bf16 in LDS.
// Block 256 thr = 8 waves; tile 64x64; each wave 16(M)x32(N); K step 32.
// ---------------------------------------------------------------------------
template<int AF32, int OBF16, int BIAS, int RES>
__global__ __launch_bounds__(256) void gemm_nn(const void* __restrict__ Av,
                                               const float* __restrict__ W,
                                               const float* __restrict__ bias,
                                               const float* __restrict__ res,
                                               void* __restrict__ Cv,
                                               int M, int N, int K) {
  __shared__ __align__(16) u16t As[64][40];   // [m][k], padded
  __shared__ __align__(16) u16t WT[64][40];   // [n][k], padded (transposed W tile)
  const int tid  = threadIdx.x;
  const int lane = tid & 31, wid = tid >> 5;
  const int wm = wid & 3, wn = wid >> 2;
  const int bm = blockIdx.y * 64, bn = blockIdx.x * 64;
  const int lm = lane & 15, lh = lane >> 4;

  f32x8 acc[2];
#pragma unroll
  for (int j = 0; j < 2; ++j)
#pragma unroll
    for (int r = 0; r < 8; ++r) acc[j][r] = 0.f;

  const int arow = tid >> 2;         // 0..63
  const int acol = (tid & 3) * 8;    // 0,8,16,24
  const int wk   = tid >> 3;         // 0..31
  const int wn0  = (tid & 7) * 8;    // 0..56

  for (int k0 = 0; k0 < K; k0 += 32) {
    if (AF32) {
      const float* A = (const float*)Av + (size_t)(bm + arow) * K + k0 + acol;
#pragma unroll
      for (int i = 0; i < 8; ++i) As[arow][acol + i] = f2bf(A[i]);
    } else {
      const u16t* A = (const u16t*)Av + (size_t)(bm + arow) * K + k0 + acol;
#if HAS_ASYNC_LDS
      ASYNC_LOAD_B128(A, &As[arow][acol]);   // 16B/thread -> ASYNCcnt
#else
      *(u16x8*)&As[arow][acol] = *(const u16x8*)A;
#endif
    }
    {
      const float* Wp = W + (size_t)(k0 + wk) * N + bn + wn0;
#pragma unroll
      for (int i = 0; i < 8; ++i) WT[wn0 + i][wk] = f2bf(Wp[i]);
      if (k0 + 32 < K) __builtin_prefetch(Wp + (size_t)32 * N, 0, 1);
    }
#if HAS_ASYNC_LDS
    if (!AF32) ASYNC_WAIT();
#endif
    __syncthreads();

    // A fragment: lane<16 -> K 0-7,16-23 ; lane>=16 -> K 8-15,24-31
    bf16x16 a = cat16(*(const u16x8*)&As[wm * 16 + lm][lh * 8],
                      *(const u16x8*)&As[wm * 16 + lm][lh * 8 + 16]);
#pragma unroll
    for (int j = 0; j < 2; ++j) {
      // B fragment: lane = N; lane<16 -> K 0-7,16-23 ; lane>=16 -> K 8-15,24-31
      bf16x16 bfr = cat16(*(const u16x8*)&WT[wn * 32 + j * 16 + lm][lh * 8],
                          *(const u16x8*)&WT[wn * 32 + j * 16 + lm][lh * 8 + 16]);
      acc[j] = __builtin_amdgcn_wmma_f32_16x16x32_bf16(false, a, false, bfr,
                                                       (short)0, acc[j], false, false);
    }
    __syncthreads();
  }

#pragma unroll
  for (int j = 0; j < 2; ++j)
#pragma unroll
    for (int r = 0; r < 8; ++r) {
      const int orow = bm + wm * 16 + lh * 8 + r;   // C layout: M = lh*8 + r
      const int ocol = bn + wn * 32 + j * 16 + lm;  //           N = lane&15
      if (orow < M && ocol < N) {
        float vacc = acc[j][r];
        if (BIAS) vacc += bias[ocol];
        if (RES)  vacc += res[(size_t)orow * N + ocol];
        if (OBF16) ((u16t*)Cv)[(size_t)orow * N + ocol] = f2bf(vacc);
        else       ((float*)Cv)[(size_t)orow * N + ocol] = vacc;
      }
    }
}

// ---------------------------------------------------------------------------
// Flash attention: grid (S/128, H, B), 8 waves/block; each wave: 16 queries.
// K tiles of 32 keys; scores & PV via bf16 WMMA; byte-mask from LDS.
// K-tile and mask-tile arrive via async global->LDS; V transposed in-regs.
// ---------------------------------------------------------------------------
__global__ __launch_bounds__(256) void attn_kernel(const u16t* __restrict__ q,
                                                   const u16t* __restrict__ k,
                                                   const u16t* __restrict__ v,
                                                   const u8t* __restrict__ mask,
                                                   u16t* __restrict__ o) {
  __shared__ __align__(16) u16t Kt[32][72];      // [key][d]
  __shared__ __align__(16) u16t VT[64][40];      // [d][key] (transposed)
  __shared__ __align__(16) u8t  Mk[128][32];     // [q_local][key_local]
  __shared__ __align__(16) u16t Pw[8][16][40];   // per-wave P transpose buffer

  const int tid = threadIdx.x;
  const int lane = tid & 31, wid = tid >> 5;
  const int lm = lane & 15, lh = lane >> 4;
  const int b = blockIdx.z, h = blockIdx.y;
  const int qblk = blockIdx.x * 128;
  const int q0 = qblk + wid * 16;
  const float scale = 0.125f;  // 1/sqrt(64)

  // Q fragments (reused over all key tiles): d 0..31 and 32..63
  const size_t qoff = ((size_t)(b * S_ + q0 + lm)) * D_ + h * HD_ + lh * 8;
  bf16x16 a0 = cat16(*(const u16x8*)(q + qoff),      *(const u16x8*)(q + qoff + 16));
  bf16x16 a1 = cat16(*(const u16x8*)(q + qoff + 32), *(const u16x8*)(q + qoff + 48));

  float mrun[8], lrun[8];
  f32x8 oacc[4];
#pragma unroll
  for (int r = 0; r < 8; ++r) { mrun[r] = -3.0e38f; lrun[r] = 0.f; }
#pragma unroll
  for (int s = 0; s < 4; ++s)
#pragma unroll
    for (int r = 0; r < 8; ++r) oacc[s][r] = 0.f;

  const int tkey = tid >> 3, td0 = (tid & 7) * 8;
  const int tqr  = tid >> 1, tc0 = (tid & 1) * 16;

  for (int kt = 0; kt < S_ / 32; ++kt) {
    const int kb = kt * 32;
    {
      const u16t* kp = k + ((size_t)(b * S_ + kb + tkey)) * D_ + h * HD_ + td0;
      const u8t* mp = mask + ((size_t)(b * S_ + qblk + tqr)) * S_ + kb + tc0;
#if HAS_ASYNC_LDS
      ASYNC_LOAD_B128(kp, &Kt[tkey][td0]);
      ASYNC_LOAD_B128(mp, &Mk[tqr][tc0]);
#else
      *(u16x8*)&Kt[tkey][td0] = *(const u16x8*)kp;
      *(uint4*)&Mk[tqr][tc0] = *(const uint4*)mp;
#endif
      const u16t* vp = v + ((size_t)(b * S_ + kb + tkey)) * D_ + h * HD_ + td0;
      u16x8 vv = *(const u16x8*)vp;
#pragma unroll
      for (int i = 0; i < 8; ++i) VT[td0 + i][tkey] = vv[i];
    }
#if HAS_ASYNC_LDS
    ASYNC_WAIT();
#endif
    __syncthreads();

    // Scores: 16 queries x 32 keys (two 16-key subtiles), K-dim = 64 (2 steps)
    f32x8 sacc[2];
#pragma unroll
    for (int j = 0; j < 2; ++j)
#pragma unroll
      for (int r = 0; r < 8; ++r) sacc[j][r] = 0.f;
#pragma unroll
    for (int j = 0; j < 2; ++j) {
      bf16x16 bk0 = cat16(*(const u16x8*)&Kt[j * 16 + lm][lh * 8],
                          *(const u16x8*)&Kt[j * 16 + lm][lh * 8 + 16]);
      sacc[j] = __builtin_amdgcn_wmma_f32_16x16x32_bf16(false, a0, false, bk0,
                                                        (short)0, sacc[j], false, false);
      bf16x16 bk1 = cat16(*(const u16x8*)&Kt[j * 16 + lm][32 + lh * 8],
                          *(const u16x8*)&Kt[j * 16 + lm][32 + lh * 8 + 16]);
      sacc[j] = __builtin_amdgcn_wmma_f32_16x16x32_bf16(false, a1, false, bk1,
                                                        (short)0, sacc[j], false, false);
    }

    // Online softmax (rows live in 16-lane halves of wave32)
    float p0[8], p1[8];
#pragma unroll
    for (int r = 0; r < 8; ++r) {
      const int qm = wid * 16 + lh * 8 + r;
      float s0v = sacc[0][r] * scale;
      float s1v = sacc[1][r] * scale;
      if (!Mk[qm][lm])      s0v = -1.0e9f;
      if (!Mk[qm][16 + lm]) s1v = -1.0e9f;
      float mx = fmaxf(s0v, s1v);
#pragma unroll
      for (int off = 1; off < 16; off <<= 1) mx = fmaxf(mx, __shfl_xor(mx, off));
      const float mnew = fmaxf(mrun[r], mx);
      const float e0 = __expf(s0v - mnew), e1 = __expf(s1v - mnew);
      float rs = e0 + e1;
#pragma unroll
      for (int off = 1; off < 16; off <<= 1) rs += __shfl_xor(rs, off);
      const float alpha = __expf(mrun[r] - mnew);
      lrun[r] = lrun[r] * alpha + rs;
      mrun[r] = mnew;
#pragma unroll
      for (int s = 0; s < 4; ++s) oacc[s][r] *= alpha;
      p0[r] = e0; p1[r] = e1;
    }

    // Transpose P (C-layout) -> A-layout via per-wave LDS (in-order DS ops)
#pragma unroll
    for (int r = 0; r < 8; ++r) {
      Pw[wid][lh * 8 + r][lm]      = f2bf(p0[r]);
      Pw[wid][lh * 8 + r][16 + lm] = f2bf(p1[r]);
    }
    bf16x16 pa = cat16(*(const u16x8*)&Pw[wid][lm][lh * 8],
                       *(const u16x8*)&Pw[wid][lm][lh * 8 + 16]);
#pragma unroll
    for (int s = 0; s < 4; ++s) {
      bf16x16 bvv = cat16(*(const u16x8*)&VT[s * 16 + lm][lh * 8],
                          *(const u16x8*)&VT[s * 16 + lm][lh * 8 + 16]);
      oacc[s] = __builtin_amdgcn_wmma_f32_16x16x32_bf16(false, pa, false, bvv,
                                                        (short)0, oacc[s], false, false);
    }
    __syncthreads();
  }

#pragma unroll
  for (int r = 0; r < 8; ++r) {
    const float inv = lrun[r] > 0.f ? 1.f / lrun[r] : 0.f;
    const int qm = q0 + lh * 8 + r;
#pragma unroll
    for (int s = 0; s < 4; ++s)
      o[((size_t)(b * S_ + qm)) * D_ + h * HD_ + s * 16 + lm] = f2bf(oacc[s][r] * inv);
  }
}

// ---------------------------------------------------------------------------
// Mask construction: one block per (b, q) row
// ---------------------------------------------------------------------------
__global__ __launch_bounds__(256) void build_masks(const int* __restrict__ node,
    const int* __restrict__ tab, const int* __restrict__ col,
    const int* __restrict__ f2p, const u8t* __restrict__ ispad,
    u8t* __restrict__ mcol, u8t* __restrict__ mfeat,
    u8t* __restrict__ mnei, u8t* __restrict__ mfull) {
  const int b = blockIdx.y, qi = blockIdx.x;
  const int base = b * S_;
  const int nq = node[base + qi], tq = tab[base + qi], cq = col[base + qi];
  const bool vq = ispad[base + qi] == 0;
  int nb[5];
#pragma unroll
  for (int i = 0; i < 5; ++i) nb[i] = f2p[(size_t)(base + qi) * 5 + i];
  const size_t row = ((size_t)(base + qi)) * S_;
  for (int kk = threadIdx.x; kk < S_; kk += 256) {
    const int nk = node[base + kk];
    const bool vk = ispad[base + kk] == 0;
    const bool pv = vq && vk;
    const bool eye = (kk == qi);
    const bool same = (nq == nk);
    bool nei = false;
#pragma unroll
    for (int i = 0; i < 5; ++i) nei = nei || (nb[i] == nk);
    mcol[row + kk]  = (u8t)(((col[base + kk] == cq) && (tab[base + kk] == tq) && pv) || eye);
    mfeat[row + kk] = (u8t)((same && pv) || eye);
    mnei[row + kk]  = (u8t)(((nei || same) && pv) || eye);
    mfull[row + kk] = (u8t)(pv || eye);
  }
}

// ---------------------------------------------------------------------------
// Fused encoder: per-token selection + RMSNorms + mask embedding + x = val+cn
// ---------------------------------------------------------------------------
__global__ __launch_bounds__(256) void encode_kernel(
    const float* __restrict__ cn_raw, const float* __restrict__ text_raw,
    const float* __restrict__ numv, const float* __restrict__ dtv, const float* __restrict__ blv,
    const float* __restrict__ W_num, const float* __restrict__ b_num,
    const float* __restrict__ W_dt, const float* __restrict__ b_dt,
    const float* __restrict__ W_bool, const float* __restrict__ b_bool,
    const float* __restrict__ g_cn, const float* __restrict__ g_num,
    const float* __restrict__ g_text, const float* __restrict__ g_dt,
    const float* __restrict__ g_bool, const float* __restrict__ mask_emb,
    const int* __restrict__ st_, const u8t* __restrict__ cellmask,
    float* __restrict__ x) {
  __shared__ float red[256];
  const int t = blockIdx.x, tid = threadIdx.x;
  const int st = st_[t];
  const u8t mk = cellmask[t];
  const float sc = (st == 0) ? numv[t] : (st == 2) ? dtv[t] : blv[t];
  const float* Wsc = (st == 0) ? W_num : (st == 2) ? W_dt : W_bool;
  const float* bsc = (st == 0) ? b_num : (st == 2) ? b_dt : b_bool;
  const float* gsel = (st == 0) ? g_num : (st == 1) ? g_text : (st == 2) ? g_dt : g_bool;

  float chosen[3], cnl[3];
  float ss_ch = 0.f, ss_cn = 0.f;
#pragma unroll
  for (int i = 0; i < 3; ++i) {
    const int d = tid + i * 256;
    float cv = (st == 1) ? text_raw[(size_t)t * D_ + d] : sc * Wsc[d] + bsc[d];
    chosen[i] = cv; ss_ch += cv * cv;
    const float c2 = cn_raw[(size_t)t * D_ + d];
    cnl[i] = c2; ss_cn += c2 * c2;
  }
  red[tid] = ss_ch; __syncthreads();
  for (int s = 128; s > 0; s >>= 1) { if (tid < s) red[tid] += red[tid + s]; __syncthreads(); }
  const float rs_ch = rsqrtf(red[0] / (float)D_ + 1e-6f);
  __syncthreads();
  red[tid] = ss_cn; __syncthreads();
  for (int s = 128; s > 0; s >>= 1) { if (tid < s) red[tid] += red[tid + s]; __syncthreads(); }
  const float rs_cn = rsqrtf(red[0] / (float)D_ + 1e-6f);

#pragma unroll
  for (int i = 0; i < 3; ++i) {
    const int d = tid + i * 256;
    float val = chosen[i] * rs_ch * gsel[d];
    if (mk) val = mask_emb[st * D_ + d];
    x[(size_t)t * D_ + d] = val + cnl[i] * rs_cn * g_cn[d];
  }
}

// ---------------------------------------------------------------------------
__global__ __launch_bounds__(256) void rmsnorm_kernel(const float* __restrict__ x,
    const float* __restrict__ g, u16t* __restrict__ out) {
  __shared__ float red[256];
  const int t = blockIdx.x, tid = threadIdx.x;
  float vals[3]; float ss = 0.f;
#pragma unroll
  for (int i = 0; i < 3; ++i) { vals[i] = x[(size_t)t * D_ + tid + i * 256]; ss += vals[i] * vals[i]; }
  red[tid] = ss; __syncthreads();
  for (int s = 128; s > 0; s >>= 1) { if (tid < s) red[tid] += red[tid + s]; __syncthreads(); }
  const float rs = rsqrtf(red[0] / (float)D_ + 1e-6f);
#pragma unroll
  for (int i = 0; i < 3; ++i) {
    const int d = tid + i * 256;
    out[(size_t)t * D_ + d] = f2bf(vals[i] * rs * g[d]);
  }
}

__global__ __launch_bounds__(256) void silu_mul_kernel(const u16t* __restrict__ gate,
    const u16t* __restrict__ up, u16t* __restrict__ h, int n) {
  for (int i = blockIdx.x * 256 + threadIdx.x; i < n; i += gridDim.x * 256) {
    const float gv = bf2f(gate[i]);
    const float uv = bf2f(up[i]);
    h[i] = f2bf((gv / (1.f + __expf(-gv))) * uv);
  }
}

__global__ __launch_bounds__(256) void dec_heads_kernel(const u16t* __restrict__ hn,
    const float* __restrict__ Wn, const float* __restrict__ bnum,
    const float* __restrict__ Wd, const float* __restrict__ bdt,
    const float* __restrict__ Wb, const float* __restrict__ bbl,
    float* __restrict__ out) {
  __shared__ float red[256];
  const int t = blockIdx.x, tid = threadIdx.x;
  float s0 = 0.f, s1 = 0.f, s2 = 0.f;
#pragma unroll
  for (int i = 0; i < 3; ++i) {
    const int d = tid + i * 256;
    const float hv = bf2f(hn[(size_t)t * D_ + d]);
    s0 += hv * Wn[d]; s1 += hv * Wd[d]; s2 += hv * Wb[d];
  }
  red[tid] = s0; __syncthreads();
  for (int s = 128; s > 0; s >>= 1) { if (tid < s) red[tid] += red[tid + s]; __syncthreads(); }
  if (tid == 0) out[t] = red[0] + bnum[0];
  __syncthreads();
  red[tid] = s1; __syncthreads();
  for (int s = 128; s > 0; s >>= 1) { if (tid < s) red[tid] += red[tid + s]; __syncthreads(); }
  if (tid == 0) out[M_ + t] = red[0] + bdt[0];
  __syncthreads();
  red[tid] = s2; __syncthreads();
  for (int s = 128; s > 0; s >>= 1) { if (tid < s) red[tid] += red[tid + s]; __syncthreads(); }
  if (tid == 0) out[2 * M_ + t] = red[0] + bbl[0];
}

// ---------------------------------------------------------------------------
extern "C" void kernel_launch(void* const* d_in, const int* in_sizes, int n_in,
                              void* d_out, int out_size, void* d_ws, size_t ws_size,
                              hipStream_t stream) {
  (void)in_sizes; (void)n_in; (void)out_size; (void)ws_size;
  const float* number_values     = (const float*)d_in[0];
  const float* datetime_values   = (const float*)d_in[1];
  const float* boolean_values    = (const float*)d_in[2];
  const float* text_values       = (const float*)d_in[3];
  const float* column_name_vals  = (const float*)d_in[4];
  const float* W_cn = (const float*)d_in[5];  const float* b_cn = (const float*)d_in[6];
  const float* W_num = (const float*)d_in[7]; const float* b_num = (const float*)d_in[8];
  const float* W_text = (const float*)d_in[9]; const float* b_text = (const float*)d_in[10];
  const float* W_dt = (const float*)d_in[11]; const float* b_dt = (const float*)d_in[12];
  const float* W_bool = (const float*)d_in[13]; const float* b_bool = (const float*)d_in[14];
  const float* g_cn = (const float*)d_in[15];
  const float* g_num = (const float*)d_in[16];
  const float* g_text = (const float*)d_in[17];
  const float* g_dt = (const float*)d_in[18];
  const float* g_bool = (const float*)d_in[19];
  const float* mask_emb = (const float*)d_in[20];
  const float* blk_norms = (const float*)d_in[21];
  const float* blk_attn = (const float*)d_in[22];
  const float* blk_up = (const float*)d_in[23];
  const float* blk_gate = (const float*)d_in[24];
  const float* blk_down = (const float*)d_in[25];
  const float* g_out = (const float*)d_in[26];
  const float* Wd_num = (const float*)d_in[27]; const float* bd_num = (const float*)d_in[28];
  const float* Wd_dt = (const float*)d_in[29];  const float* bd_dt = (const float*)d_in[30];
  const float* Wd_bool = (const float*)d_in[31]; const float* bd_bool = (const float*)d_in[32];
  const float* Wd_text = (const float*)d_in[33]; const float* bd_text = (const float*)d_in[34];
  const int* node_indices = (const int*)d_in[35];
  const int* table_name_indices = (const int*)d_in[36];
  const int* column_name_indices = (const int*)d_in[37];
  const int* f2p = (const int*)d_in[38];
  const int* semantic_types = (const int*)d_in[39];
  const u8t* cellmasks = (const u8t*)d_in[40];
  const u8t* is_padding = (const u8t*)d_in[41];

  char* ws = (char*)d_ws;
  size_t off = 0;
  auto alloc = [&](size_t bytes) -> void* {
    void* p = ws + off; off = (off + bytes + 255) & ~(size_t)255; return p;
  };
  float* x  = (float*)alloc((size_t)M_ * D_ * 4);
  u16t* xn  = (u16t*)alloc((size_t)M_ * D_ * 2);
  u16t* qb  = (u16t*)alloc((size_t)M_ * D_ * 2);
  u16t* kb  = (u16t*)alloc((size_t)M_ * D_ * 2);
  u16t* vb  = (u16t*)alloc((size_t)M_ * D_ * 2);
  u16t* ob  = (u16t*)alloc((size_t)M_ * D_ * 2);
  u16t* gate = (u16t*)alloc((size_t)M_ * DFF_ * 2);
  u16t* up   = (u16t*)alloc((size_t)M_ * DFF_ * 2);
  u8t* mplanes = (u8t*)alloc((size_t)4 * B_ * S_ * S_);
  // Encoder scratch aliases the MLP buffers (used strictly before them)
  float* cn_raw   = (float*)gate;
  float* text_raw = (float*)up;

  u8t* mcol  = mplanes;
  u8t* mfeat = mplanes + (size_t)1 * B_ * S_ * S_;
  u8t* mnei  = mplanes + (size_t)2 * B_ * S_ * S_;
  u8t* mfull = mplanes + (size_t)3 * B_ * S_ * S_;
  const u8t* mlist[4] = {mcol, mfeat, mnei, mfull};

  build_masks<<<dim3(S_, B_), 256, 0, stream>>>(node_indices, table_name_indices,
      column_name_indices, f2p, is_padding, mcol, mfeat, mnei, mfull);

  // Encoder GEMMs (fp32 A -> bf16 in LDS), bias fused
  gemm_nn<1, 0, 1, 0><<<dim3(D_ / 64, M_ / 64), 256, 0, stream>>>(
      column_name_vals, W_cn, b_cn, nullptr, cn_raw, M_, D_, DT_);
  gemm_nn<1, 0, 1, 0><<<dim3(D_ / 64, M_ / 64), 256, 0, stream>>>(
      text_values, W_text, b_text, nullptr, text_raw, M_, D_, DT_);
  encode_kernel<<<M_, 256, 0, stream>>>(cn_raw, text_raw,
      number_values, datetime_values, boolean_values,
      W_num, b_num, W_dt, b_dt, W_bool, b_bool,
      g_cn, g_num, g_text, g_dt, g_bool, mask_emb,
      semantic_types, cellmasks, x);

  for (int i = 0; i < L_; ++i) {
    for (int j = 0; j < 4; ++j) {
      const float* nw = blk_norms + ((size_t)i * 5 + j) * D_;
      rmsnorm_kernel<<<M_, 256, 0, stream>>>(x, nw, xn);
      const float* aw = blk_attn + (((size_t)i * 4 + j) * 4) * D_ * D_;
      gemm_nn<0, 1, 0, 0><<<dim3(D_ / 64, M_ / 64), 256, 0, stream>>>(
          xn, aw + (size_t)0 * D_ * D_, nullptr, nullptr, qb, M_, D_, D_);
      gemm_nn<0, 1, 0, 0><<<dim3(D_ / 64, M_ / 64), 256, 0, stream>>>(
          xn, aw + (size_t)1 * D_ * D_, nullptr, nullptr, kb, M_, D_, D_);
      gemm_nn<0, 1, 0, 0><<<dim3(D_ / 64, M_ / 64), 256, 0, stream>>>(
          xn, aw + (size_t)2 * D_ * D_, nullptr, nullptr, vb, M_, D_, D_);
      attn_kernel<<<dim3(S_ / 128, H_, B_), 256, 0, stream>>>(qb, kb, vb, mlist[j], ob);
      gemm_nn<0, 0, 0, 1><<<dim3(D_ / 64, M_ / 64), 256, 0, stream>>>(
          ob, aw + (size_t)3 * D_ * D_, nullptr, x, x, M_, D_, D_);
    }
    const float* nw4 = blk_norms + ((size_t)i * 5 + 4) * D_;
    rmsnorm_kernel<<<M_, 256, 0, stream>>>(x, nw4, xn);
    gemm_nn<0, 1, 0, 0><<<dim3(DFF_ / 64, M_ / 64), 256, 0, stream>>>(
        xn, blk_gate + (size_t)i * D_ * DFF_, nullptr, nullptr, gate, M_, DFF_, D_);
    gemm_nn<0, 1, 0, 0><<<dim3(DFF_ / 64, M_ / 64), 256, 0, stream>>>(
        xn, blk_up + (size_t)i * D_ * DFF_, nullptr, nullptr, up, M_, DFF_, D_);
    silu_mul_kernel<<<1024, 256, 0, stream>>>(gate, up, gate, M_ * DFF_);
    gemm_nn<0, 0, 0, 1><<<dim3(D_ / 64, M_ / 64), 256, 0, stream>>>(
        gate, blk_down + (size_t)i * DFF_ * D_, nullptr, x, x, M_, D_, DFF_);
  }

  rmsnorm_kernel<<<M_, 256, 0, stream>>>(x, g_out, xn);
  float* outf = (float*)d_out;
  dec_heads_kernel<<<M_, 256, 0, stream>>>(xn, Wd_num, bd_num, Wd_dt, bd_dt,
                                           Wd_bool, bd_bool, outf);
  gemm_nn<0, 0, 1, 0><<<dim3(DT_ / 64, M_ / 64), 256, 0, stream>>>(
      xn, Wd_text, bd_text, nullptr, outf + (size_t)3 * M_, M_, DT_, D_);
}
